// MoEMLP_32504312496610
// MI455X (gfx1250) — matrix operand order repaired and checked by reading, not compiled, not run on hardware
//
#include <hip/hip_runtime.h>
#include <hip/hip_bf16.h>
#include <math.h>

typedef __attribute__((ext_vector_type(16))) __bf16       v16bf;
typedef __attribute__((ext_vector_type(8)))  float        v8f;
typedef __attribute__((ext_vector_type(4)))  unsigned int u32x4;
typedef __attribute__((ext_vector_type(8)))  int          i32x8;
typedef __attribute__((ext_vector_type(4)))  int          i32x4;

#define NT    4096      // B*T tokens
#define HDIM  1024
#define DFFD  4096
#define NE    8
#define MAXS  4096
#define BM    16        // token rows per block
#define FK    128       // DFF chunk width

// LDS strides (elements). Chosen so row*stride*2B is 16B aligned and the
// dword-stride % 64 == 4 -> 16 rows hit 16 distinct banks.
#define XST   1032      // X hi/lo planes: 1032*2B = 2064 = 16*129
#define HST   136       // H hi/lo planes: 136*2B = 272 = 16*17

// LDS layout (bytes from dynamic-shared base)
#define OFF_XH  0
#define OFF_XL  (OFF_XH + BM * XST * 2)            // 33024
#define OFF_HH  (OFF_XL + BM * XST * 2)            // 66048
#define OFF_HL  (OFF_HH + BM * HST * 2)            // 70400
#define OFF_W   (OFF_HL + BM * HST * 2)            // 74752 (16B aligned)
#define W_REGION_BYTES (32 * 1024 * 4)             // 128KB (W2 tile; W1 dbl-buf aliased)
#define OFF_TS  (OFF_W + W_REGION_BYTES)           // 205824
#define OFF_GS  (OFF_TS + 64)
#define SMEM_BYTES (OFF_GS + 64)                   // 205952

#define W1TILE_ELEMS (32 * FK)                     // 4096 floats = 16KB

__device__ __forceinline__ float gelu_tanh(float x) {
    const float c0 = 0.7978845608028654f;   // sqrt(2/pi)
    const float c1 = 0.044715f;
    float u = c0 * (x + c1 * x * x * x);
#if __has_builtin(__builtin_amdgcn_tanhf)
    float t = __builtin_amdgcn_tanhf(u);    // v_tanh_f32, straight-line
#else
    float au = fabsf(u);
    float e2 = __expf(-2.0f * au);
    float t  = (1.0f - e2) / (1.0f + e2);
    t = copysignf(t, u);
#endif
    return 0.5f * x * (1.0f + t);
}

// ---- Tensor Data Mover: 2D tile (h rows x w fp32 elems, row stride in elems) -> LDS ----
__device__ __forceinline__ void tdm_load_2d(unsigned lds_addr, const void* gptr,
                                            unsigned w, unsigned h, unsigned stride) {
    u32x4 g0;
    i32x8 g1;
    i32x4 g2 = {};
    i32x4 g3 = {};
    unsigned long long ga = (unsigned long long)gptr;
    g0.x = 1u;                                             // count=1, user mode
    g0.y = lds_addr;                                       // LDS byte address
    g0.z = (unsigned)ga;                                   // global_addr[31:0]
    g0.w = (unsigned)((ga >> 32) & 0x01FFFFFFull)          // global_addr[56:32]
         | 0x80000000u;                                    // type=2 ("image")
    g1[0] = 0x00020000;                                    // data_size=4B, no mask/pad/iter
    g1[1] = (int)((w & 0xFFFFu) << 16);                    // tensor_dim0[15:0]
    g1[2] = (int)((w >> 16) | ((h & 0xFFFFu) << 16));      // tensor_dim0 hi | tensor_dim1 lo
    g1[3] = (int)((h >> 16) | (w << 16));                  // tensor_dim1 hi | tile_dim0=w
    g1[4] = (int)h;                                        // tile_dim1=h, tile_dim2=0
    g1[5] = (int)stride;                                   // tensor_dim0_stride lo
    g1[6] = 0;
    g1[7] = 0;
#if defined(__clang_major__) && (__clang_major__ >= 23)
    i32x8 g4 = {};
    __builtin_amdgcn_tensor_load_to_lds(g0, g1, g2, g3, g4, 0);
#else
    __builtin_amdgcn_tensor_load_to_lds(g0, g1, g2, g3, 0);
#endif
}

// ---------------- Kernel 1: router + top-2 + scatter to expert lists ----------------
__global__ void router_topk_kernel(const float* __restrict__ X,
                                   const float* __restrict__ RW,
                                   int* __restrict__ cnt,
                                   int* __restrict__ list,
                                   float* __restrict__ gate) {
    int wid  = threadIdx.x >> 5;
    int lane = threadIdx.x & 31;
    int t = blockIdx.x * 8 + wid;           // one wave32 per token

    float acc[NE];
#pragma unroll
    for (int e = 0; e < NE; ++e) acc[e] = 0.0f;

    const float* xr = X + (size_t)t * HDIM;
    for (int j = lane; j < HDIM; j += 32) {
        float xv = xr[j];
#pragma unroll
        for (int e = 0; e < NE; ++e) acc[e] += xv * RW[e * HDIM + j];
    }
#pragma unroll
    for (int e = 0; e < NE; ++e) {
#pragma unroll
        for (int off = 16; off > 0; off >>= 1)
            acc[e] += __shfl_xor(acc[e], off, 32);
    }

    if (lane == 0) {
        int i0 = 0; float m0 = acc[0];
#pragma unroll
        for (int e = 1; e < NE; ++e) if (acc[e] > m0) { m0 = acc[e]; i0 = e; }
        int i1 = -1; float m1 = -3.4e38f;
#pragma unroll
        for (int e = 0; e < NE; ++e) if (e != i0 && acc[e] > m1) { m1 = acc[e]; i1 = e; }
        float p1 = __expf(m1 - m0);         // stable: m0 >= m1
        float inv = 1.0f / (1.0f + p1);
        float w0 = inv, w1 = p1 * inv;

        int s0 = atomicAdd(&cnt[i0], 1);
        list[i0 * MAXS + s0] = t;  gate[i0 * MAXS + s0] = w0;
        int s1 = atomicAdd(&cnt[i1], 1);
        list[i1 * MAXS + s1] = t;  gate[i1 * MAXS + s1] = w1;
    }
}

// ---------------- Kernel 2: fused expert MLP ----------------
__global__ __launch_bounds__(256) void moe_expert_kernel(
    const float* __restrict__ X,
    const float* __restrict__ W1, const float* __restrict__ B1,
    const float* __restrict__ W2, const float* __restrict__ B2,
    const int* __restrict__ cnt, const int* __restrict__ list,
    const float* __restrict__ gate, float* __restrict__ out)
{
    extern __shared__ char smem[];
    __bf16* XsHi = (__bf16*)(smem + OFF_XH);     // BM x XST
    __bf16* XsLo = (__bf16*)(smem + OFF_XL);
    __bf16* HsHi = (__bf16*)(smem + OFF_HH);     // BM x HST
    __bf16* HsLo = (__bf16*)(smem + OFF_HL);
    float*  Wreg = (float*)(smem + OFF_W);       // weight tile region (TDM dest)
    int*    Ts   = (int*)(smem + OFF_TS);
    float*  Gs   = (float*)(smem + OFF_GS);

    const unsigned ldsBase = (unsigned)(unsigned long long)(smem);
    const unsigned ldsW0   = ldsBase + OFF_W;                      // W1 buf 0
    const unsigned ldsW1   = ldsBase + OFF_W + W1TILE_ELEMS * 4;   // W1 buf 1
    const unsigned ldsW2   = ldsBase + OFF_W;                      // W2 tile (whole region)

    const int e    = blockIdx.y;
    const int n    = cnt[e];
    const int base = blockIdx.x * BM;
    if (base >= n) return;                       // uniform across block

    const int tid = threadIdx.x;
    if (tid < BM) {
        int slot = base + tid;
        if (slot < n) { Ts[tid] = list[e * MAXS + slot]; Gs[tid] = gate[e * MAXS + slot]; }
        else          { Ts[tid] = -1;                    Gs[tid] = 0.0f; }
    }
    __syncthreads();

    // Gather 16 token rows; pre-split fp32 -> bf16 hi/lo planes in LDS.
    for (int idx = tid; idx < BM * HDIM; idx += 256) {
        int r = idx >> 10;                 // HDIM == 1024
        int c = idx & (HDIM - 1);
        int tok = Ts[r];
        float xv = (tok >= 0) ? X[(size_t)tok * HDIM + c] : 0.0f;
        __bf16 h = (__bf16)xv;
        XsHi[r * XST + c] = h;
        XsLo[r * XST + c] = (__bf16)(xv - (float)h);
    }
    __syncthreads();

    const int wid     = tid >> 5;          // 0..7
    const int lane    = tid & 31;
    const int halfsel = lane >> 4;
    const int lrow    = lane & 15;
    const int kbA     = halfsel * 16;      // A-fragment K-group base
    const int kbB     = halfsel * 16;      // B-fragment K base

    const float* W1e = W1 + (size_t)e * HDIM * DFFD;
    const float* W2e = W2 + (size_t)e * DFFD * HDIM;

    union Frag { v16bf v; u32x4 q[2]; };

    v8f zero = {};
    v8f accN[8];                           // 16 rows x 128 cols per wave
#pragma unroll
    for (int i = 0; i < 8; ++i) accN[i] = zero;

#pragma unroll 1
    for (int cf = 0; cf < DFFD / FK; ++cf) {
        const int chunkBase = cf * FK;

        // ---- Phase 1: h[16 x FK] = gelu(X.W1 + b1); W1 tiles via TDM, double-buffered ----
        if (wid == 0) {
            tdm_load_2d(ldsW0, W1e + chunkBase, FK, 32, DFFD);
            __builtin_amdgcn_s_wait_tensorcnt(0);
        }
        __syncthreads();

        v8f hacc = zero;
        const int col1 = chunkBase + wid * 16 + lrow;
        int cur = 0;
#pragma unroll 1
        for (int k = 0; k < HDIM; k += 32) {
            if (wid == 0 && (k + 32) < HDIM) {          // prefetch next tile (async)
                tdm_load_2d(cur ? ldsW0 : ldsW1,
                            W1e + (size_t)(k + 32) * DFFD + chunkBase, FK, 32, DFFD);
            }
            const float* Wc = Wreg + cur * W1TILE_ELEMS;

            Frag ah, al;                                 // A from pre-split LDS planes
            const __bf16* ph = XsHi + lrow * XST + k + kbA;
            const __bf16* pl = XsLo + lrow * XST + k + kbA;
            ah.q[0] = *(const u32x4*)(ph);
            ah.q[1] = *(const u32x4*)(ph + 8);
            al.q[0] = *(const u32x4*)(pl);
            al.q[1] = *(const u32x4*)(pl + 8);

            v16bf bhi, blo;                              // B from TDM tile (hi/lo split)
#pragma unroll
            for (int j = 0; j < 16; ++j) {
                float wv = Wc[(kbB + j) * FK + wid * 16 + lrow];
                __bf16 h = (__bf16)wv;
                bhi[j] = h;
                blo[j] = (__bf16)(wv - (float)h);
            }
            hacc = __builtin_amdgcn_wmma_f32_16x16x32_bf16(false, ah.v, false, bhi, (short)0, hacc, false, false);
            hacc = __builtin_amdgcn_wmma_f32_16x16x32_bf16(false, ah.v, false, blo, (short)0, hacc, false, false);
            hacc = __builtin_amdgcn_wmma_f32_16x16x32_bf16(false, al.v, false, bhi, (short)0, hacc, false, false);

            if (wid == 0) __builtin_amdgcn_s_wait_tensorcnt(0);
            __syncthreads();
            cur ^= 1;
        }
        {
            float bias1 = B1[e * DFFD + col1];
#pragma unroll
            for (int i = 0; i < 8; ++i) {
                float hv = gelu_tanh(hacc[i] + bias1);
                int r = i + halfsel * 8;                 // C layout: M = vgpr + 8*half
                __bf16 hh = (__bf16)hv;
                HsHi[r * HST + wid * 16 + lrow] = hh;
                HsLo[r * HST + wid * 16 + lrow] = (__bf16)(hv - (float)hh);
            }
        }
        __syncthreads();

        // ---- Phase 2: out[16 x 1024] += h_chunk . W2[chunk rows]; W2 tiles via TDM ----
#pragma unroll 1
        for (int k2 = 0; k2 < FK; k2 += 32) {
            if (wid == 0) {
                tdm_load_2d(ldsW2, W2e + (size_t)(chunkBase + k2) * HDIM, HDIM, 32, HDIM);
                __builtin_amdgcn_s_wait_tensorcnt(0);
            }
            __syncthreads();

            Frag ah, al;
            const __bf16* ph = HsHi + lrow * HST + k2 + kbA;
            const __bf16* pl = HsLo + lrow * HST + k2 + kbA;
            ah.q[0] = *(const u32x4*)(ph);
            ah.q[1] = *(const u32x4*)(ph + 8);
            al.q[0] = *(const u32x4*)(pl);
            al.q[1] = *(const u32x4*)(pl + 8);

#pragma unroll
            for (int nt = 0; nt < 8; ++nt) {
                int col2 = wid * 128 + nt * 16 + lrow;
                v16bf bhi, blo;
#pragma unroll
                for (int j = 0; j < 16; ++j) {
                    float wv = Wreg[(kbB + j) * HDIM + col2];
                    __bf16 h = (__bf16)wv;
                    bhi[j] = h;
                    blo[j] = (__bf16)(wv - (float)h);
                }
                accN[nt] = __builtin_amdgcn_wmma_f32_16x16x32_bf16(false, ah.v, false, bhi, (short)0, accN[nt], false, false);
                accN[nt] = __builtin_amdgcn_wmma_f32_16x16x32_bf16(false, ah.v, false, blo, (short)0, accN[nt], false, false);
                accN[nt] = __builtin_amdgcn_wmma_f32_16x16x32_bf16(false, al.v, false, bhi, (short)0, accN[nt], false, false);
            }
            __syncthreads();   // before next TDM overwrites Wreg
        }
    }

    // ---- Epilogue: bias2, gate, scatter-add into output (2 experts per token) ----
#pragma unroll
    for (int nt = 0; nt < 8; ++nt) {
        int col = wid * 128 + nt * 16 + lrow;
        float bias2 = B2[e * HDIM + col];
#pragma unroll
        for (int i = 0; i < 8; ++i) {
            int r = i + halfsel * 8;
            int slot = base + r;
            if (slot < n) {
                int tok = Ts[r];
                atomicAdd(&out[(size_t)tok * HDIM + col], Gs[r] * (accN[nt][i] + bias2));
            }
        }
    }
}

// ---------------- Launch ----------------
extern "C" void kernel_launch(void* const* d_in, const int* in_sizes, int n_in,
                              void* d_out, int out_size, void* d_ws, size_t ws_size,
                              hipStream_t stream) {
    (void)in_sizes; (void)n_in; (void)ws_size;
    const float* X  = (const float*)d_in[0];   // [B,T,H]
    const float* RW = (const float*)d_in[1];   // [E,H]
    const float* W1 = (const float*)d_in[2];   // [E,H,DFF]
    const float* B1 = (const float*)d_in[3];   // [E,DFF]
    const float* W2 = (const float*)d_in[4];   // [E,DFF,H]
    const float* B2 = (const float*)d_in[5];   // [E,H]
    float* out = (float*)d_out;

    int*   cnt  = (int*)d_ws;
    int*   list = (int*)((char*)d_ws + 256);
    float* gate = (float*)((char*)d_ws + 256 + (size_t)NE * MAXS * 4);

    hipMemsetAsync(d_ws, 0, 256, stream);
    hipMemsetAsync(d_out, 0, (size_t)out_size * sizeof(float), stream);

    router_topk_kernel<<<NT / 8, 256, 0, stream>>>(X, RW, cnt, list, gate);

    dim3 grid(MAXS / BM, NE);
    moe_expert_kernel<<<grid, 256, SMEM_BYTES, stream>>>(X, W1, B1, W2, B2,
                                                         cnt, list, gate, out);
}
